// MultiHeadAttention_37263136260720
// MI455X (gfx1250) — matrix operand order repaired and checked by reading, not compiled, not run on hardware
//
#include <hip/hip_runtime.h>

// Problem constants (from the reference): L=2048, B=2, D=1024, H=16, DK=64
#define L_SEQ 2048
#define BATCH 2
#define DMODEL 1024
#define NHEAD 16
#define DKH 64
#define BH (BATCH * NHEAD)
#define MROWS (L_SEQ * BATCH)
#define QSCALE 0.125f  // DK^-0.5

typedef __attribute__((ext_vector_type(16))) _Float16 v16h;
typedef __attribute__((ext_vector_type(8)))  float    v8f;

// Async global->LDS copy support (gfx1250: GLOBAL_LOAD_ASYNC_TO_LDS_B128,
// tracked with ASYNCcnt / s_wait_asynccnt).
#if defined(__has_builtin)
#if __has_builtin(__builtin_amdgcn_global_load_async_to_lds_b128) && \
    __has_builtin(__builtin_amdgcn_s_wait_asynccnt)
#define HAVE_ASYNC_LDS 1
#endif
#endif
#ifndef HAVE_ASYNC_LDS
#define HAVE_ASYNC_LDS 0
#endif

#if HAVE_ASYNC_LDS
typedef int v4i __attribute__((vector_size(16)));
typedef __attribute__((address_space(1))) v4i* gv4i_ptr;
typedef __attribute__((address_space(3))) v4i* lv4i_ptr;
__device__ __forceinline__ void async_copy_b128(const void* gsrc, void* lds) {
  __builtin_amdgcn_global_load_async_to_lds_b128(
      (gv4i_ptr)gsrc, (lv4i_ptr)lds, /*offset=*/0, /*cpol=*/0);
}
#endif

// ---------------------------------------------------------------------------
// WMMA fragment element maps (CDNA5 ISA 7.12.2, wave32):
//  A (16x32 f16): lane%16 = M row. lanes 0-15: e0..7 -> K=e, e8..15 -> K=e+8;
//                 lanes 16-31: +8.
//  B (32x16 f16): lane%16 = N col. same K map as A (e0..7 -> K=e, e8..15 -> K=e+8; lanes 16-31: +8).
//  C/D (16x16 f32): element e -> M = e + 8*(lane>=16), N = lane%16.
// ---------------------------------------------------------------------------
__device__ __forceinline__ int amap(int e, int hi) {
  return (e & 7) + ((e >> 3) << 4) + (hi << 3);
}
__device__ __forceinline__ int bmap(int e, int hi) {
  return (e & 7) + ((e >> 3) << 4) + (hi << 3);
}

__device__ __forceinline__ v16h a_frag_f32(const float* __restrict__ row, int hi) {
  v16h r;
#pragma unroll
  for (int e = 0; e < 16; ++e) r[e] = (_Float16)row[amap(e, hi)];
  return r;
}
__device__ __forceinline__ v16h b_frag_f32(const float* __restrict__ row, int hi) {
  v16h r;
#pragma unroll
  for (int e = 0; e < 16; ++e) r[e] = (_Float16)row[bmap(e, hi)];
  return r;
}
__device__ __forceinline__ v16h a_frag_f16(const _Float16* row, int hi) {
  v16h r;
#pragma unroll
  for (int e = 0; e < 16; ++e) r[e] = row[amap(e, hi)];
  return r;
}
__device__ __forceinline__ v16h b_frag_f16(const _Float16* row, int hi) {
  v16h r;
#pragma unroll
  for (int e = 0; e < 16; ++e) r[e] = row[bmap(e, hi)];
  return r;
}

__device__ __forceinline__ v8f wmma_f16(v16h a, v16h b, v8f c) {
  // v_wmma_f32_16x16x32_f16: D = A(16x32) x B(32x16) + C, f32 accumulate
  return __builtin_amdgcn_wmma_f32_16x16x32_f16(false, a, false, b, (short)0, c,
                                                false, false);
}

// ---------------------------------------------------------------------------
// Projection GEMM: out = (X[M,K] @ W[N,K]^T + bias) * scale, written f16.
// transposedV==0: head-major [B*H, L, DKH]   (Q, K)
// transposedV==1: head-major [B*H, DKH, L]   (V^T, so attention B-fragments
//                                             become row-contiguous loads)
// Wave tile 64x64, block tile 128x256 (8 waves).
// ---------------------------------------------------------------------------
__global__ __launch_bounds__(256) void proj_qkv_kernel(
    const float* __restrict__ X, const float* __restrict__ W,
    const float* __restrict__ bias, _Float16* __restrict__ out_hm, float scale,
    int transposedV) {
  const int lane = threadIdx.x & 31;
  const int wave = threadIdx.x >> 5;
  const int hi = lane >> 4;
  const int ln = lane & 15;
  const int m0 = blockIdx.y * 128 + (wave & 1) * 64;
  const int n0 = blockIdx.x * 256 + (wave >> 1) * 64;

  v8f acc[4][4] = {};

  for (int k0 = 0; k0 < DMODEL; k0 += 32) {
    if (k0 + 128 < DMODEL) {
#pragma unroll
      for (int i = 0; i < 4; ++i)
        __builtin_prefetch(X + (size_t)(m0 + i * 16 + ln) * DMODEL + k0 + 128, 0, 1);
    }
    v16h af[4], bf[4];
#pragma unroll
    for (int i = 0; i < 4; ++i)
      af[i] = a_frag_f32(X + (size_t)(m0 + i * 16 + ln) * DMODEL + k0, hi);
#pragma unroll
    for (int j = 0; j < 4; ++j)
      bf[j] = b_frag_f32(W + (size_t)(n0 + j * 16 + ln) * DMODEL + k0, hi);
#pragma unroll
    for (int i = 0; i < 4; ++i)
#pragma unroll
      for (int j = 0; j < 4; ++j) acc[i][j] = wmma_f16(af[i], bf[j], acc[i][j]);
  }

#pragma unroll
  for (int i = 0; i < 4; ++i) {
#pragma unroll
    for (int j = 0; j < 4; ++j) {
      const int n = n0 + j * 16 + ln;
      const int h = n >> 6;   // n / DKH
      const int dk = n & 63;  // n % DKH
      const float bn = bias[n];
#pragma unroll
      for (int e = 0; e < 8; ++e) {
        const int m = m0 + i * 16 + e + (hi << 3);
        const int l = m >> 1;  // m / BATCH
        const int b = m & 1;   // m % BATCH
        const _Float16 val = (_Float16)((acc[i][j][e] + bn) * scale);
        if (transposedV)
          out_hm[((size_t)(b * NHEAD + h) * DKH + dk) * L_SEQ + l] = val;
        else
          out_hm[((size_t)(b * NHEAD + h) * L_SEQ + l) * DKH + dk] = val;
      }
    }
  }
}

// ---------------------------------------------------------------------------
// Flash attention: one wave handles a 32-row query tile for one (b,h).
// K/V tiles are double-buffered in LDS via GLOBAL_LOAD_ASYNC_TO_LDS_B128
// (copy of tile t+1 overlaps WMMA work on tile t; completion via ASYNCcnt).
// Online softmax over S in steps of 32; probs reshaped via LDS into
// A-fragments for the P@V WMMAs. V is consumed transposed ([B*H, DKH, L]).
// Writes f16 [M, D] with m = l*B + b.
// ---------------------------------------------------------------------------
__global__ __launch_bounds__(32) void flash_attn_kernel(
    const _Float16* __restrict__ q16, const _Float16* __restrict__ k16,
    const _Float16* __restrict__ vT16, _Float16* __restrict__ attn16) {
  __shared__ _Float16 lds_p[2 * 16 * 32];
#if HAVE_ASYNC_LDS
  __shared__ _Float16 lds_k[2][32 * 64];  // [buf][s_local][dk]
  __shared__ _Float16 lds_v[2][64 * 32];  // [buf][dk][s_local]
#endif
  const int lane = threadIdx.x & 31;
  const int hi = lane >> 4;
  const int ln = lane & 15;
  const int bh = blockIdx.y;
  const int l0 = blockIdx.x * 32;

  const _Float16* __restrict__ qb = q16 + ((size_t)bh * L_SEQ + l0) * DKH;
  const _Float16* __restrict__ kb = k16 + (size_t)bh * L_SEQ * DKH;
  const _Float16* __restrict__ vtb = vT16 + (size_t)bh * DKH * L_SEQ;

#if HAVE_ASYNC_LDS
  // Issue one tile's async copies: 8x b128 for K (contiguous 4KB block) and
  // 8x b128 for V^T (64 rows x 64B). 16 ASYNCcnt increments per tile.
  auto issue_tile = [&](int buf, int s0) {
    const char* gk = (const char*)(kb + (size_t)s0 * DKH);
    char* lk = (char*)&lds_k[buf][0];
#pragma unroll
    for (int c = 0; c < 8; ++c)
      async_copy_b128(gk + c * 512 + lane * 16, lk + c * 512 + lane * 16);
#pragma unroll
    for (int c = 0; c < 8; ++c) {
      const int row = c * 8 + (lane >> 2);
      const int off = (lane & 3) * 16;
      async_copy_b128((const char*)(vtb + (size_t)row * L_SEQ + s0) + off,
                      (char*)&lds_v[buf][row * 32] + off);
    }
  };
  issue_tile(0, 0);
#endif

  v16h qf[2][2];
#pragma unroll
  for (int mi = 0; mi < 2; ++mi)
#pragma unroll
    for (int c = 0; c < 2; ++c)
      qf[mi][c] = a_frag_f16(qb + (size_t)(mi * 16 + ln) * DKH + c * 32, hi);

  float mrun[2][8], lrun[2][8];
  v8f acc[2][4] = {};
#pragma unroll
  for (int mi = 0; mi < 2; ++mi)
#pragma unroll
    for (int e = 0; e < 8; ++e) { mrun[mi][e] = -3.0e38f; lrun[mi][e] = 0.0f; }

  for (int s0 = 0; s0 < L_SEQ; s0 += 32) {
#if HAVE_ASYNC_LDS
    const int buf = (s0 >> 5) & 1;
    if (s0 + 32 < L_SEQ) {
      issue_tile(buf ^ 1, s0 + 32);            // prefetch next tile
      __builtin_amdgcn_s_wait_asynccnt(16);    // current tile's 16 ops done
    } else {
      __builtin_amdgcn_s_wait_asynccnt(0);
    }
#endif

    // K fragments shared by both query row-tiles
    v16h kf[2][2];
#pragma unroll
    for (int j = 0; j < 2; ++j)
#pragma unroll
      for (int c2 = 0; c2 < 2; ++c2)
#if HAVE_ASYNC_LDS
        kf[j][c2] = b_frag_f16(&lds_k[buf][(j * 16 + ln) * DKH + c2 * 32], hi);
#else
        kf[j][c2] = b_frag_f16(kb + (size_t)(s0 + j * 16 + ln) * DKH + c2 * 32, hi);
#endif

    // scores: 2 (query tiles) x 2 (key tiles) of 16x16, K=64 in two x32 WMMAs
    v8f sc[2][2];
#pragma unroll
    for (int mi = 0; mi < 2; ++mi)
#pragma unroll
      for (int j = 0; j < 2; ++j) {
        v8f c = {};
#pragma unroll
        for (int c2 = 0; c2 < 2; ++c2) c = wmma_f16(qf[mi][c2], kf[j][c2], c);
        sc[mi][j] = c;
      }

    // online softmax per query tile
#pragma unroll
    for (int mi = 0; mi < 2; ++mi) {
      float newm[8];
#pragma unroll
      for (int e = 0; e < 8; ++e) {
        float v = fmaxf(sc[mi][0][e], sc[mi][1][e]);
#pragma unroll
        for (int off = 1; off < 16; off <<= 1) v = fmaxf(v, __shfl_xor(v, off, 32));
        newm[e] = fmaxf(mrun[mi][e], v);
        const float corr = __expf(mrun[mi][e] - newm[e]);
        mrun[mi][e] = newm[e];
        lrun[mi][e] *= corr;
#pragma unroll
        for (int f = 0; f < 4; ++f) acc[mi][f][e] *= corr;
      }
#pragma unroll
      for (int j = 0; j < 2; ++j) {
#pragma unroll
        for (int e = 0; e < 8; ++e) {
          float p = __expf(sc[mi][j][e] - newm[e]);
          float ps = p;
#pragma unroll
          for (int off = 1; off < 16; off <<= 1) ps += __shfl_xor(ps, off, 32);
          lrun[mi][e] += ps;
          // D-fragment (row = e+8*hi, col = j*16+ln) -> LDS row-major 16x32
          lds_p[mi * 512 + (e + (hi << 3)) * 32 + j * 16 + ln] = (_Float16)p;
        }
      }
    }
    __syncthreads();

    // V^T fragments: element e -> s_local = bmap(e,hi) (AMD table order), col = dk
    v16h vf[4];
#pragma unroll
    for (int f = 0; f < 4; ++f)
#if HAVE_ASYNC_LDS
      vf[f] = b_frag_f16(&lds_v[buf][(f * 16 + ln) * 32], hi);
#else
      vf[f] = b_frag_f16(vtb + (size_t)(f * 16 + ln) * L_SEQ + s0, hi);
#endif

#pragma unroll
    for (int mi = 0; mi < 2; ++mi) {
      v16h pf = a_frag_f16(&lds_p[mi * 512 + ln * 32], hi);
#pragma unroll
      for (int f = 0; f < 4; ++f) acc[mi][f] = wmma_f16(pf, vf[f], acc[mi][f]);
    }
    __syncthreads();
  }

  const int b = bh >> 4;           // bh / NHEAD
  const int h = bh & (NHEAD - 1);  // bh % NHEAD
#pragma unroll
  for (int mi = 0; mi < 2; ++mi) {
#pragma unroll
    for (int f = 0; f < 4; ++f) {
#pragma unroll
      for (int e = 0; e < 8; ++e) {
        const int l = l0 + mi * 16 + e + (hi << 3);
        const size_t m = (size_t)l * BATCH + b;
        const int n = h * DKH + f * 16 + ln;
        attn16[m * DMODEL + n] = (_Float16)(acc[mi][f][e] / lrun[mi][e]);
      }
    }
  }
}

// ---------------------------------------------------------------------------
// Output projection: d_out[M,N] = A16[M,K] @ WO[N,K]^T + bias (f32 out).
// ---------------------------------------------------------------------------
__global__ __launch_bounds__(256) void out_proj_kernel(
    const _Float16* __restrict__ A, const float* __restrict__ W,
    const float* __restrict__ bias, float* __restrict__ out) {
  const int lane = threadIdx.x & 31;
  const int wave = threadIdx.x >> 5;
  const int hi = lane >> 4;
  const int ln = lane & 15;
  const int m0 = blockIdx.y * 128 + (wave & 1) * 64;
  const int n0 = blockIdx.x * 256 + (wave >> 1) * 64;

  v8f acc[4][4] = {};

  for (int k0 = 0; k0 < DMODEL; k0 += 32) {
    if (k0 + 128 < DMODEL) {
#pragma unroll
      for (int i = 0; i < 4; ++i)
        __builtin_prefetch(A + (size_t)(m0 + i * 16 + ln) * DMODEL + k0 + 128, 0, 1);
    }
    v16h af[4], bf[4];
#pragma unroll
    for (int i = 0; i < 4; ++i)
      af[i] = a_frag_f16(A + (size_t)(m0 + i * 16 + ln) * DMODEL + k0, hi);
#pragma unroll
    for (int j = 0; j < 4; ++j)
      bf[j] = b_frag_f32(W + (size_t)(n0 + j * 16 + ln) * DMODEL + k0, hi);
#pragma unroll
    for (int i = 0; i < 4; ++i)
#pragma unroll
      for (int j = 0; j < 4; ++j) acc[i][j] = wmma_f16(af[i], bf[j], acc[i][j]);
  }

#pragma unroll
  for (int i = 0; i < 4; ++i) {
#pragma unroll
    for (int j = 0; j < 4; ++j) {
      const int n = n0 + j * 16 + ln;
      const float bn = bias[n];
#pragma unroll
      for (int e = 0; e < 8; ++e) {
        const int m = m0 + i * 16 + e + (hi << 3);
        out[(size_t)m * DMODEL + n] = acc[i][j][e] + bn;
      }
    }
  }
}

// ---------------------------------------------------------------------------
extern "C" void kernel_launch(void* const* d_in, const int* in_sizes, int n_in,
                              void* d_out, int out_size, void* d_ws, size_t ws_size,
                              hipStream_t stream) {
  (void)in_sizes; (void)n_in; (void)out_size; (void)ws_size;
  const float* query = (const float*)d_in[0];
  const float* key_  = (const float*)d_in[1];
  const float* value = (const float*)d_in[2];
  const float* WQ_w  = (const float*)d_in[3];
  const float* WQ_b  = (const float*)d_in[4];
  const float* WK_w  = (const float*)d_in[5];
  const float* WK_b  = (const float*)d_in[6];
  const float* WV_w  = (const float*)d_in[7];
  const float* WV_b  = (const float*)d_in[8];
  const float* WO_w  = (const float*)d_in[9];
  const float* WO_b  = (const float*)d_in[10];

  const size_t headElems = (size_t)BH * L_SEQ * DKH;  // 4M f16 each
  _Float16* q16  = (_Float16*)d_ws;
  _Float16* k16  = q16 + headElems;
  _Float16* vT16 = k16 + headElems;
  _Float16* a16  = vT16 + headElems;

  dim3 gemmGrid(DMODEL / 256, MROWS / 128);  // (4, 32)
  proj_qkv_kernel<<<gemmGrid, 256, 0, stream>>>(query, WQ_w, WQ_b, q16, QSCALE, 0);
  proj_qkv_kernel<<<gemmGrid, 256, 0, stream>>>(key_,  WK_w, WK_b, k16, 1.0f, 0);
  proj_qkv_kernel<<<gemmGrid, 256, 0, stream>>>(value, WV_w, WV_b, vT16, 1.0f, 1);

  dim3 attnGrid(L_SEQ / 32, BH);  // (64, 32)
  flash_attn_kernel<<<attnGrid, 32, 0, stream>>>(q16, k16, vT16, a16);

  out_proj_kernel<<<gemmGrid, 256, 0, stream>>>(a16, WO_w, WO_b, (float*)d_out);
}
